// NeuralPCFGSrcParser_88338887344489
// MI455X (gfx1250) — matrix operand order repaired and checked by reading, not compiled, not run on hardware
//
#include <hip/hip_runtime.h>

#define NEGV (-1000000000.0f)

namespace {
constexpr int Bz   = 64;      // batch
constexpr int Nn   = 32;      // tokens
constexpr int Vv   = 50257;   // vocab
constexpr int DIMc = 256;
constexpr int NTc  = 30;
constexpr int PTc  = 60;
constexpr int Sc   = 90;      // NT+PT
constexpr int SSc  = 8100;    // S*S
constexpr int KBLK = 254;     // K blocks of 32
constexpr int KPAD = KBLK * 32; // 8128
constexpr int MAXSPANS = Bz * (Nn - 1); // 1984 spans at w=2
}

typedef __attribute__((ext_vector_type(8)))  float        v8f;
typedef __attribute__((ext_vector_type(16))) __bf16       v16bf;
typedef __attribute__((ext_vector_type(4)))  unsigned int u32x4;

#if __has_builtin(__builtin_amdgcn_global_load_async_to_lds_b32) && \
    __has_builtin(__builtin_amdgcn_s_wait_asynccnt)
#define USE_ASYNC_LDS 1
#endif

__device__ __forceinline__ size_t choff(int w, int b, int i, int s) {
  return (((size_t)w * Bz + b) * Nn + i) * Sc + s;
}

// round-to-nearest-even f32 -> bf16 (as raw u16)
__device__ __forceinline__ unsigned short f2bf(float f) {
  union { float f; unsigned u; } t; t.f = f;
  unsigned r = t.u + 0x7FFFu + ((t.u >> 16) & 1u);
  return (unsigned short)(r >> 16);
}

#ifdef USE_ASYNC_LDS
// async global->LDS copy of one dword (tracked by ASYNCcnt).
// Low 32 bits of a generic LDS pointer are the LDS offset on AMDGPU.
// NOTE: builtin takes a NON-const AS(1) pointer.
__device__ __forceinline__ void async_g2l_b32(const float* g, float* l) {
  __attribute__((address_space(1))) int* gp =
      (__attribute__((address_space(1))) int*)(uintptr_t)g;
  __attribute__((address_space(3))) int* lp =
      (__attribute__((address_space(3))) int*)(unsigned int)(uintptr_t)l;
  __builtin_amdgcn_global_load_async_to_lds_b32(gp, lp, 0, 0);
}
#endif

// B-pack fragment index for rules matrix: B[k][a], k in [0,8128), a in [0,32)
__device__ __forceinline__ size_t bpack_idx(int a, int k) {
  int kb = k >> 5, kk = k & 31;
  int h = kk >> 4, j = kk & 15;
  int lane = h * 16 + (a & 15);
  int ntile = a >> 4;
  return ((((size_t)ntile * KBLK + kb) * 32) + lane) * 16 + j;
}

// ---------------------------------------------------------------- init chart
__global__ void k_init_chart(float* chart, size_t total) {
  size_t i = (size_t)blockIdx.x * blockDim.x + threadIdx.x;
  if (i < total) chart[i] = NEGV;
}

__global__ void k_zero_bpack(unsigned short* bp, size_t total) {
  size_t i = (size_t)blockIdx.x * blockDim.x + threadIdx.x;
  if (i < total) bp[i] = 0;
}

// ------------------------------------------------- rules: softmax + pack bf16
__global__ void k_rules(const float* __restrict__ nt_emb,
                        const float* __restrict__ rule_w,
                        const float* __restrict__ rule_b,
                        float* __restrict__ rlog,
                        unsigned short* __restrict__ Bpack) {
  __shared__ float emb[DIMc];
  __shared__ float red[256];
  int a = blockIdx.x, tid = threadIdx.x;
  emb[tid] = nt_emb[a * DIMc + tid];
  __syncthreads();

  float lmax = -3.0e38f;
  for (int k = tid; k < SSc; k += 256) {
    float acc = rule_b[k];
#pragma unroll 8
    for (int d = 0; d < DIMc; ++d) acc = fmaf(emb[d], rule_w[(size_t)d * SSc + k], acc);
    rlog[(size_t)a * SSc + k] = acc;
    lmax = fmaxf(lmax, acc);
  }
  red[tid] = lmax; __syncthreads();
  for (int o = 128; o > 0; o >>= 1) { if (tid < o) red[tid] = fmaxf(red[tid], red[tid + o]); __syncthreads(); }
  float rowmax = red[0]; __syncthreads();

  float lsum = 0.f;
  for (int k = tid; k < SSc; k += 256) lsum += __expf(rlog[(size_t)a * SSc + k] - rowmax);
  red[tid] = lsum; __syncthreads();
  for (int o = 128; o > 0; o >>= 1) { if (tid < o) red[tid] += red[tid + o]; __syncthreads(); }
  float inv = 1.0f / red[0];

  for (int k = tid; k < SSc; k += 256) {
    float v = __expf(rlog[(size_t)a * SSc + k] - rowmax) * inv;
    Bpack[bpack_idx(a, k)] = f2bf(v);
  }
}

// ----------------------------------------------------------- root MLP + lsm
__global__ void k_root(const float* __restrict__ root_emb,
                       const float* __restrict__ in_w, const float* __restrict__ in_b,
                       const float* __restrict__ w1a, const float* __restrict__ b1a,
                       const float* __restrict__ w2a, const float* __restrict__ b2a,
                       const float* __restrict__ w1b, const float* __restrict__ b1b,
                       const float* __restrict__ w2b, const float* __restrict__ b2b,
                       const float* __restrict__ out_w, const float* __restrict__ out_b,
                       float* __restrict__ roots) {
  __shared__ float h[DIMc], t[DIMc], r[NTc];
  __shared__ float lse;
  int j = threadIdx.x;
  h[j] = root_emb[j];
  __syncthreads();
  float acc = in_b[j];
#pragma unroll 8
  for (int d = 0; d < DIMc; ++d) acc = fmaf(h[d], in_w[d * DIMc + j], acc);
  __syncthreads(); h[j] = acc; __syncthreads();

  const float* W1[2] = {w1a, w1b}; const float* Bb1[2] = {b1a, b1b};
  const float* W2[2] = {w2a, w2b}; const float* Bb2[2] = {b2a, b2b};
  for (int L = 0; L < 2; ++L) {
    acc = Bb1[L][j];
#pragma unroll 8
    for (int d = 0; d < DIMc; ++d) acc = fmaf(h[d], W1[L][d * DIMc + j], acc);
    t[j] = fmaxf(acc, 0.f);
    __syncthreads();
    acc = Bb2[L][j];
#pragma unroll 8
    for (int d = 0; d < DIMc; ++d) acc = fmaf(t[d], W2[L][d * DIMc + j], acc);
    acc = fmaxf(acc, 0.f);
    __syncthreads(); h[j] += acc; __syncthreads();
  }
  if (j < NTc) {
    acc = out_b[j];
#pragma unroll 8
    for (int d = 0; d < DIMc; ++d) acc = fmaf(h[d], out_w[d * NTc + j], acc);
    r[j] = acc;
  }
  __syncthreads();
  if (j == 0) {
    float M = -3.0e38f;
    for (int a = 0; a < NTc; ++a) M = fmaxf(M, r[a]);
    float Z = 0.f;
    for (int a = 0; a < NTc; ++a) Z += __expf(r[a] - M);
    lse = M + __logf(Z);
  }
  __syncthreads();
  if (j < NTc) roots[j] = r[j] - lse;
}

// ----------------------------------------------------- vocab MLP hidden (60)
__global__ void k_vocab_hidden(const float* __restrict__ pt_emb,
                               const float* __restrict__ in_w, const float* __restrict__ in_b,
                               const float* __restrict__ w1a, const float* __restrict__ b1a,
                               const float* __restrict__ w2a, const float* __restrict__ b2a,
                               const float* __restrict__ w1b, const float* __restrict__ b1b,
                               const float* __restrict__ w2b, const float* __restrict__ b2b,
                               float* __restrict__ hv) {
  __shared__ float h[DIMc], t[DIMc];
  int j = threadIdx.x, p = blockIdx.x;
  h[j] = pt_emb[p * DIMc + j];
  __syncthreads();
  float acc = in_b[j];
#pragma unroll 8
  for (int d = 0; d < DIMc; ++d) acc = fmaf(h[d], in_w[d * DIMc + j], acc);
  __syncthreads(); h[j] = acc; __syncthreads();

  const float* W1[2] = {w1a, w1b}; const float* Bb1[2] = {b1a, b1b};
  const float* W2[2] = {w2a, w2b}; const float* Bb2[2] = {b2a, b2b};
  for (int L = 0; L < 2; ++L) {
    acc = Bb1[L][j];
#pragma unroll 8
    for (int d = 0; d < DIMc; ++d) acc = fmaf(h[d], W1[L][d * DIMc + j], acc);
    t[j] = fmaxf(acc, 0.f);
    __syncthreads();
    acc = Bb2[L][j];
#pragma unroll 8
    for (int d = 0; d < DIMc; ++d) acc = fmaf(t[d], W2[L][d * DIMc + j], acc);
    acc = fmaxf(acc, 0.f);
    __syncthreads(); h[j] += acc; __syncthreads();
  }
  hv[p * DIMc + j] = h[j];
}

// ----------------------- vocab logits: per-(row,chunk) partial max & sumexp
__global__ void k_vocab_norm(const float* __restrict__ hv,
                             const float* __restrict__ out_w,
                             const float* __restrict__ out_b,
                             float* __restrict__ partials) {
  __shared__ float h[DIMc];
  __shared__ float red[256];
  int p = blockIdx.x, chunk = blockIdx.y, tid = threadIdx.x;
  h[tid] = hv[p * DIMc + tid];
  __syncthreads();
  float vals[32];
  float lmax = -3.0e38f;
  int base = chunk * 8192 + tid;
#pragma unroll
  for (int q = 0; q < 32; ++q) {
    int v = base + q * 256;
    float a = -3.0e38f;
    if (v < Vv) {
      a = out_b[v];
#pragma unroll 8
      for (int d = 0; d < DIMc; ++d) a = fmaf(h[d], out_w[(size_t)d * Vv + v], a);
    }
    vals[q] = a;
    lmax = fmaxf(lmax, a);
  }
  red[tid] = lmax; __syncthreads();
  for (int o = 128; o > 0; o >>= 1) { if (tid < o) red[tid] = fmaxf(red[tid], red[tid + o]); __syncthreads(); }
  float gmax = red[0]; __syncthreads();
  float lsum = 0.f;
#pragma unroll
  for (int q = 0; q < 32; ++q) {
    int v = base + q * 256;
    if (v < Vv) lsum += __expf(vals[q] - gmax);
  }
  red[tid] = lsum; __syncthreads();
  for (int o = 128; o > 0; o >>= 1) { if (tid < o) red[tid] += red[tid + o]; __syncthreads(); }
  if (tid == 0) {
    partials[((size_t)p * 7 + chunk) * 2 + 0] = gmax;
    partials[((size_t)p * 7 + chunk) * 2 + 1] = red[0];
  }
}

__global__ void k_vocab_combine(const float* __restrict__ partials, float* __restrict__ logZ) {
  int p = threadIdx.x;
  if (p < PTc) {
    float M = -3.0e38f;
    for (int c = 0; c < 7; ++c) M = fmaxf(M, partials[((size_t)p * 7 + c) * 2 + 0]);
    float Z = 0.f;
    for (int c = 0; c < 7; ++c)
      Z += partials[((size_t)p * 7 + c) * 2 + 1] * __expf(partials[((size_t)p * 7 + c) * 2 + 0] - M);
    logZ[p] = M + __logf(Z);
  }
}

// ------------- terms: chart[1][b][n][NT+p] = logit(p, x[b][n]) - logZ[p]
__global__ void k_terms(const float* __restrict__ hv,
                        const float* __restrict__ out_w,
                        const float* __restrict__ out_b,
                        const float* __restrict__ logZ,
                        const int* __restrict__ x,
                        float* __restrict__ chart) {
  int b = blockIdx.x / Nn, n = blockIdx.x % Nn;
  int tok = x[b * Nn + n];
  int p = threadIdx.x;
  if (p < PTc) {
    float acc = 0.f;
#pragma unroll 8
    for (int d = 0; d < DIMc; ++d) acc = fmaf(hv[p * DIMc + d], out_w[(size_t)d * Vv + tok], acc);
    chart[choff(1, b, n, NTc + p)] = acc + out_b[tok] - logZ[p];
  }
}

// ------------------------------- phase 1: per-span E = sum_u outer(expL, expR)
__global__ void k_span_exp(const float* __restrict__ chart,
                           unsigned short* __restrict__ Ebuf,
                           float* __restrict__ Mspan,
                           int w) {
  __shared__ float Lx[31][Sc];
  __shared__ float Rx[31][Sc];
  __shared__ float maxLu[31];
  __shared__ float maxRu[31];
  __shared__ float Msh;
  int tid = threadIdx.x;
  int span = blockIdx.x;
  int nsp = (Nn + 1) - w;
  int b = span / nsp, i = span % nsp;
  int nu = w - 1;

#ifdef USE_ASYNC_LDS
  // CDNA5 async global->LDS: no VGPR staging, tracked by ASYNCcnt.
  for (int t0 = tid; t0 < nu * Sc; t0 += 256) {
    int u = t0 / Sc, s = t0 % Sc;
    async_g2l_b32(&chart[choff(u + 1, b, i, s)], &Lx[u][s]);
    async_g2l_b32(&chart[choff(w - (u + 1), b, i + u + 1, s)], &Rx[u][s]);
  }
  __builtin_amdgcn_s_wait_asynccnt(0);
  __syncthreads();
#else
  for (int t0 = tid; t0 < nu * Sc; t0 += 256) {
    int u = t0 / Sc, s = t0 % Sc;
    Lx[u][s] = chart[choff(u + 1, b, i, s)];
    Rx[u][s] = chart[choff(w - (u + 1), b, i + u + 1, s)];
  }
  __syncthreads();
#endif

  if (tid < nu) {
    float m = -3.0e38f;
    for (int s = 0; s < Sc; ++s) m = fmaxf(m, Lx[tid][s]);
    maxLu[tid] = m;
  } else if (tid < 2 * nu) {
    int u = tid - nu;
    float m = -3.0e38f;
    for (int s = 0; s < Sc; ++s) m = fmaxf(m, Rx[u][s]);
    maxRu[u] = m;
  }
  __syncthreads();
  if (tid == 0) {
    float M = -3.0e38f;
    for (int u = 0; u < nu; ++u) M = fmaxf(M, maxLu[u] + maxRu[u]);
    Msh = M;
    Mspan[span] = M;
  }
  __syncthreads();
  float M = Msh;
  for (int t0 = tid; t0 < nu * Sc; t0 += 256) {
    int u = t0 / Sc, s = t0 % Sc;
    Lx[u][s] = __expf(Lx[u][s] - maxLu[u]);          // <= 1
    Rx[u][s] = __expf(Rx[u][s] - (M - maxLu[u]));    // <= 1
  }
  __syncthreads();
  unsigned short* erow = Ebuf + (size_t)span * KPAD;
  for (int k = tid; k < KPAD; k += 256) {
    float acc = 0.f;
    if (k < SSc) {
      int s = k / Sc, t = k % Sc;
      for (int u = 0; u < nu; ++u) acc = fmaf(Lx[u][s], Rx[u][t], acc);
    }
    erow[k] = f2bf(acc);
  }
}

// --------------- phase 2: WMMA GEMM  Y(spans x 32) = E(spans x 8128) @ B(8128 x 32)
// block = 64 threads (2 waves); each wave owns TWO 16x16 M-tiles sharing one
// B fragment (halves B traffic, 2 independent WMMAs per K-step). grid.y = 2 N-tiles.
__global__ void k_span_gemm(const unsigned short* __restrict__ Ebuf,
                            const unsigned short* __restrict__ Bpack,
                            const float* __restrict__ Mspan,
                            float* __restrict__ chart,
                            int w) {
  int lane = threadIdx.x & 31;
  int wv   = threadIdx.x >> 5;          // 0..1
  int pair  = blockIdx.x * 2 + wv;      // pair of M-tiles; pairs = 2*nsp
  int ntile = blockIdx.y;
  int h = lane >> 4, col = lane & 15;
  int mt0 = pair * 2, mt1 = pair * 2 + 1;

  const unsigned short* a0row = Ebuf + (size_t)(mt0 * 16 + col) * KPAD;
  const unsigned short* a1row = Ebuf + (size_t)(mt1 * 16 + col) * KPAD;
  v8f c0 = {0.f, 0.f, 0.f, 0.f, 0.f, 0.f, 0.f, 0.f};
  v8f c1 = {0.f, 0.f, 0.f, 0.f, 0.f, 0.f, 0.f, 0.f};

#pragma unroll 2
  for (int kb = 0; kb < KBLK; ++kb) {
    // A k order used here (contiguous, not the ISA table order): lane holds K h*16..h*16+15 (matches Bpack)
    union { u32x4 u[2]; v16bf v; } A0, A1, Bm;
    const unsigned short* bp = Bpack + ((((size_t)ntile * KBLK + kb) * 32) + lane) * 16;
    Bm.u[0] = *reinterpret_cast<const u32x4*>(bp);
    Bm.u[1] = *reinterpret_cast<const u32x4*>(bp + 8);
    A0.u[0] = *reinterpret_cast<const u32x4*>(a0row + kb * 32 + h * 16);
    A0.u[1] = *reinterpret_cast<const u32x4*>(a0row + kb * 32 + h * 16 + 8);
    A1.u[0] = *reinterpret_cast<const u32x4*>(a1row + kb * 32 + h * 16);
    A1.u[1] = *reinterpret_cast<const u32x4*>(a1row + kb * 32 + h * 16 + 8);
    if (kb + 1 < KBLK) {
      __builtin_prefetch((const void*)(bp + 32 * 16), 0, 3);          // reused: near
      __builtin_prefetch((const void*)(a0row + (kb + 1) * 32), 0, 0); // streamed: far
      __builtin_prefetch((const void*)(a1row + (kb + 1) * 32), 0, 0);
    }
    c0 = __builtin_amdgcn_wmma_f32_16x16x32_bf16(false, A0.v, false, Bm.v,
                                                 (short)0, c0, false, false);
    c1 = __builtin_amdgcn_wmma_f32_16x16x32_bf16(false, A1.v, false, Bm.v,
                                                 (short)0, c1, false, false);
  }

  int nsp = (Nn + 1) - w;
  int nt = ntile * 16 + col;
  if (nt < NTc) {
#pragma unroll
    for (int r = 0; r < 8; ++r) {
      int m = h * 8 + r;
      {
        int span = mt0 * 16 + m;
        float y = c0[r];
        float beta = (y > 0.f) ? (__logf(y) + Mspan[span]) : NEGV;
        int b = span / nsp, i = span % nsp;
        chart[choff(w, b, i, nt)] = beta;
      }
      {
        int span = mt1 * 16 + m;
        float y = c1[r];
        float beta = (y > 0.f) ? (__logf(y) + Mspan[span]) : NEGV;
        int b = span / nsp, i = span % nsp;
        chart[choff(w, b, i, nt)] = beta;
      }
    }
  }
}

// ----------------------------------------------- final: logsumexp(top + roots)
__global__ void k_final(const float* __restrict__ chart,
                        const float* __restrict__ roots,
                        const int* __restrict__ lengths,
                        float* __restrict__ out) {
  int b = blockIdx.x, a = threadIdx.x;
  int len = lengths[b];
  float v = (a < NTc) ? (chart[choff(len, b, 0, a)] + roots[a]) : -3.0e38f;
  float m = v;
  for (int o = 16; o > 0; o >>= 1) m = fmaxf(m, __shfl_xor(m, o, 32));
  float e = (a < NTc) ? __expf(v - m) : 0.f;
  float s = e;
  for (int o = 16; o > 0; o >>= 1) s += __shfl_xor(s, o, 32);
  if (a == 0) out[b] = m + __logf(s);
}

// ---------------------------------------------------------------------------
extern "C" void kernel_launch(void* const* d_in, const int* in_sizes, int n_in,
                              void* d_out, int out_size, void* d_ws, size_t ws_size,
                              hipStream_t stream) {
  // params flattened in JAX pytree order (sorted dict keys, list order):
  const float* nt_emb   = (const float*)d_in[0];   // (30,256)
  const float* pt_emb   = (const float*)d_in[1];   // (60,256)
  const float* root_emb = (const float*)d_in[2];   // (1,256)
  const float* r_b1a = (const float*)d_in[3];
  const float* r_b2a = (const float*)d_in[4];
  const float* r_w1a = (const float*)d_in[5];
  const float* r_w2a = (const float*)d_in[6];
  const float* r_b1b = (const float*)d_in[7];
  const float* r_b2b = (const float*)d_in[8];
  const float* r_w1b = (const float*)d_in[9];
  const float* r_w2b = (const float*)d_in[10];
  const float* r_inb = (const float*)d_in[11];
  const float* r_inw = (const float*)d_in[12];
  const float* r_outb = (const float*)d_in[13];    // (30,)
  const float* r_outw = (const float*)d_in[14];    // (256,30)
  const float* rule_b = (const float*)d_in[15];    // (8100,)
  const float* rule_w = (const float*)d_in[16];    // (256,8100)
  const float* v_b1a = (const float*)d_in[17];
  const float* v_b2a = (const float*)d_in[18];
  const float* v_w1a = (const float*)d_in[19];
  const float* v_w2a = (const float*)d_in[20];
  const float* v_b1b = (const float*)d_in[21];
  const float* v_b2b = (const float*)d_in[22];
  const float* v_w1b = (const float*)d_in[23];
  const float* v_w2b = (const float*)d_in[24];
  const float* v_inb = (const float*)d_in[25];
  const float* v_inw = (const float*)d_in[26];
  const float* v_outb = (const float*)d_in[27];    // (50257,)
  const float* v_outw = (const float*)d_in[28];    // (256,50257)
  const int*   x       = (const int*)d_in[29];     // (64,32)
  const int*   lengths = (const int*)d_in[30];     // (64,)

  // workspace carve
  size_t off = 0;
  auto alloc = [&](size_t bytes) -> void* {
    void* r = (void*)((char*)d_ws + off);
    off += (bytes + 255) & ~(size_t)255;
    return r;
  };
  const size_t chart_elems = (size_t)(Nn + 1) * Bz * Nn * Sc; // 6,082,560
  float*          chart    = (float*)alloc(chart_elems * 4);
  unsigned short* Bpack    = (unsigned short*)alloc((size_t)2 * KBLK * 32 * 16 * 2);
  float*          rlog     = (float*)alloc((size_t)NTc * SSc * 4);
  float*          hv       = (float*)alloc((size_t)PTc * DIMc * 4);
  float*          partials = (float*)alloc((size_t)PTc * 7 * 2 * 4);
  float*          logZ     = (float*)alloc(64 * 4);
  float*          rootsv   = (float*)alloc(32 * 4);
  float*          Mspan    = (float*)alloc((size_t)MAXSPANS * 4);
  unsigned short* Ebuf     = (unsigned short*)alloc((size_t)MAXSPANS * KPAD * 2);
  (void)ws_size; (void)n_in; (void)in_sizes; (void)out_size;

  // init
  k_init_chart<<<(unsigned)((chart_elems + 255) / 256), 256, 0, stream>>>(chart, chart_elems);
  const size_t bpe = (size_t)2 * KBLK * 32 * 16;
  k_zero_bpack<<<(unsigned)((bpe + 255) / 256), 256, 0, stream>>>(Bpack, bpe);

  // rules softmax -> packed bf16 B matrix
  k_rules<<<NTc, 256, 0, stream>>>(nt_emb, rule_w, rule_b, rlog, Bpack);

  // root MLP -> log-softmax roots
  k_root<<<1, 256, 0, stream>>>(root_emb, r_inw, r_inb, r_w1a, r_b1a, r_w2a, r_b2a,
                                r_w1b, r_b1b, r_w2b, r_b2b, r_outw, r_outb, rootsv);

  // vocab MLP -> hidden, normalizers, terms into chart[1]
  k_vocab_hidden<<<PTc, 256, 0, stream>>>(pt_emb, v_inw, v_inb, v_w1a, v_b1a, v_w2a, v_b2a,
                                          v_w1b, v_b1b, v_w2b, v_b2b, hv);
  k_vocab_norm<<<dim3(PTc, 7), 256, 0, stream>>>(hv, v_outw, v_outb, partials);
  k_vocab_combine<<<1, 64, 0, stream>>>(partials, logZ);
  k_terms<<<Bz * Nn, 64, 0, stream>>>(hv, v_outw, v_outb, logZ, x, chart);

  // inside loop
  for (int w = 2; w <= Nn; ++w) {
    int nsp = (Nn + 1) - w;
    int spans = Bz * nsp;           // multiple of 64 -> mtiles = 4*nsp, pairs = 2*nsp
    k_span_exp<<<spans, 256, 0, stream>>>(chart, Ebuf, Mspan, w);
    dim3 g(nsp, 2);                 // 2 waves/block, 2 M-tile pairs/block
    k_span_gemm<<<g, 64, 0, stream>>>(Ebuf, Bpack, Mspan, chart, w);
  }

  k_final<<<Bz, 32, 0, stream>>>(chart, rootsv, lengths, (float*)d_out);
}
